// RoPEAttention_67164698575257
// MI455X (gfx1250) — matrix operand order repaired and checked by reading, not compiled, not run on hardware
//
#include <hip/hip_runtime.h>

// ---------------------------------------------------------------------------
// RoPE multi-head attention for MI455X (gfx1250), bf16 WMMA path.
//   B=4, T=2048, HID=2048, NH=16, HD=128
// Flash attention stages K/V tiles into LDS with GLOBAL_LOAD_ASYNC_TO_LDS_B128
// (ASYNCcnt) shared by all 8 waves of a block (8x global-traffic reduction).
// ---------------------------------------------------------------------------

typedef __attribute__((ext_vector_type(16))) __bf16 v16bf;
typedef __attribute__((ext_vector_type(8)))  __bf16 bf16x8;
typedef __attribute__((ext_vector_type(8)))  float  v8f;

static constexpr int Bc   = 4;
static constexpr int Tc   = 2048;
static constexpr int HIDc = 2048;
static constexpr int NHc  = 16;
static constexpr int HDc  = 128;
static constexpr int Mrows = Bc * Tc;          // 8192
static constexpr float QSCALE = 0.08838834764831845f; // 1/sqrt(128)

// ---- wave32 16-lane xor reductions via ds_swizzle (group-of-32 xor mode) ----
__device__ __forceinline__ float lane16_max(float x) {
    x = fmaxf(x, __int_as_float(__builtin_amdgcn_ds_swizzle(__float_as_int(x), 0x041f)));
    x = fmaxf(x, __int_as_float(__builtin_amdgcn_ds_swizzle(__float_as_int(x), 0x081f)));
    x = fmaxf(x, __int_as_float(__builtin_amdgcn_ds_swizzle(__float_as_int(x), 0x101f)));
    x = fmaxf(x, __int_as_float(__builtin_amdgcn_ds_swizzle(__float_as_int(x), 0x201f)));
    return x;
}
__device__ __forceinline__ float lane16_sum(float x) {
    x += __int_as_float(__builtin_amdgcn_ds_swizzle(__float_as_int(x), 0x041f));
    x += __int_as_float(__builtin_amdgcn_ds_swizzle(__float_as_int(x), 0x081f));
    x += __int_as_float(__builtin_amdgcn_ds_swizzle(__float_as_int(x), 0x101f));
    x += __int_as_float(__builtin_amdgcn_ds_swizzle(__float_as_int(x), 0x201f));
    return x;
}

__device__ __forceinline__ v16bf pack16(bf16x8 lo, bf16x8 hi) {
    v16bf r;
#pragma unroll
    for (int i = 0; i < 8; ++i) { r[i] = lo[i]; r[i + 8] = hi[i]; }
    return r;
}

// ---- CDNA5 async memory->LDS copy (16 bytes per lane), tracked on ASYNCcnt ----
__device__ __forceinline__ void async_ld16(unsigned lds_byte_off, const void* gptr) {
    asm volatile("global_load_async_to_lds_b128 %0, %1, off"
                 :: "v"(lds_byte_off),
                    "v"((unsigned long long)(uintptr_t)gptr)
                 : "memory");
}
__device__ __forceinline__ void async_wait0() {
    asm volatile("s_wait_asynccnt 0" ::: "memory");
}

// ---------------------------------------------------------------------------
// Tiled GEMM: C[M,N] = A[M,K] * W[N,K]^T   (A,W f32 in memory, bf16 compute)
// Block tile 64(M) x 128(N) x 32(K); 8 waves, each wave computes 32x32.
// MODE 0: C = A*W^T + bias  -> f32 Cout (row-major M x N)
// MODE 1: qkv scatter into bf16 qb[b,h,t,d], kb[b,h,t,d], vt[b,h,d,t]
// ---------------------------------------------------------------------------
template <int MODE>
__global__ __launch_bounds__(256)
void gemm_bf16_wmma(const float* __restrict__ A, const float* __restrict__ W,
                    const float* __restrict__ bias, float* __restrict__ Cout,
                    __bf16* __restrict__ qb, __bf16* __restrict__ kb,
                    __bf16* __restrict__ vt,
                    int M, int N, int K) {
    __shared__ alignas(16) __bf16 As[64][32];
    __shared__ alignas(16) __bf16 Bs[128][32];

    const int tid  = threadIdx.x;
    const int lane = tid & 31;
    const int wid  = tid >> 5;
    const int wm   = wid & 1;        // 0..1 -> M offset wm*32
    const int wn   = wid >> 1;       // 0..3 -> N offset wn*32
    const int half = lane >> 4;
    const int lm   = lane & 15;

    const int m0 = blockIdx.y * 64;
    const int n0 = blockIdx.x * 128;

    v8f acc[2][2] = {};

    for (int kk = 0; kk < K; kk += 32) {
        // ---- cooperative load A tile: 64x32 (8 f32 per thread) ----
        {
            const int idx = tid * 8;
            const int r = idx >> 5, c = idx & 31;
            const float* src = A + (size_t)(m0 + r) * K + kk + c;
            float4 f0 = *(const float4*)(src);
            float4 f1 = *(const float4*)(src + 4);
            bf16x8 v;
            v[0]=(__bf16)f0.x; v[1]=(__bf16)f0.y; v[2]=(__bf16)f0.z; v[3]=(__bf16)f0.w;
            v[4]=(__bf16)f1.x; v[5]=(__bf16)f1.y; v[6]=(__bf16)f1.z; v[7]=(__bf16)f1.w;
            *(bf16x8*)&As[r][c] = v;
        }
        // ---- cooperative load W tile: 128x32 (16 f32 per thread) ----
        {
            const int idx = tid * 16;
            const int r = idx >> 5, c = idx & 31;
            const float* src = W + (size_t)(n0 + r) * K + kk + c;
            float4 f0 = *(const float4*)(src);
            float4 f1 = *(const float4*)(src + 4);
            float4 f2 = *(const float4*)(src + 8);
            float4 f3 = *(const float4*)(src + 12);
            bf16x8 v0, v1;
            v0[0]=(__bf16)f0.x; v0[1]=(__bf16)f0.y; v0[2]=(__bf16)f0.z; v0[3]=(__bf16)f0.w;
            v0[4]=(__bf16)f1.x; v0[5]=(__bf16)f1.y; v0[6]=(__bf16)f1.z; v0[7]=(__bf16)f1.w;
            v1[0]=(__bf16)f2.x; v1[1]=(__bf16)f2.y; v1[2]=(__bf16)f2.z; v1[3]=(__bf16)f2.w;
            v1[4]=(__bf16)f3.x; v1[5]=(__bf16)f3.y; v1[6]=(__bf16)f3.z; v1[7]=(__bf16)f3.w;
            *(bf16x8*)&Bs[r][c]     = v0;
            *(bf16x8*)&Bs[r][c + 8] = v1;
        }
        __syncthreads();

        // ---- build fragments (ISA 16-bit A 16x32 / B 32x16 layouts) ----
        v16bf afr[2], bfr[2];
#pragma unroll
        for (int im = 0; im < 2; ++im) {
            const int row = wm * 32 + im * 16 + lm;
            bf16x8 lo = *(const bf16x8*)&As[row][half * 8];
            bf16x8 hi = *(const bf16x8*)&As[row][16 + half * 8];
            afr[im] = pack16(lo, hi);
        }
#pragma unroll
        for (int in_ = 0; in_ < 2; ++in_) {
            const int col = wn * 32 + in_ * 16 + lm;
            bf16x8 lo = *(const bf16x8*)&Bs[col][half * 8];
            bf16x8 hi = *(const bf16x8*)&Bs[col][16 + half * 8];
            bfr[in_] = pack16(lo, hi);
        }
#pragma unroll
        for (int im = 0; im < 2; ++im)
#pragma unroll
            for (int in_ = 0; in_ < 2; ++in_)
                acc[im][in_] = __builtin_amdgcn_wmma_f32_16x16x32_bf16(
                    false, afr[im], false, bfr[in_], (short)0, acc[im][in_],
                    false, false);
        __syncthreads();
    }

    // ---- epilogue: C 16x16 f32 layout: VGPR r -> M=r (lanes 0-15), r+8 (16-31)
#pragma unroll
    for (int im = 0; im < 2; ++im) {
#pragma unroll
        for (int in_ = 0; in_ < 2; ++in_) {
#pragma unroll
            for (int r = 0; r < 8; ++r) {
                const int m = m0 + wm * 32 + im * 16 + r + 8 * half;
                const int n = n0 + wn * 32 + in_ * 16 + lm;
                const float val = acc[im][in_][r];
                if constexpr (MODE == 0) {
                    Cout[(size_t)m * N + n] = val + bias[n];
                } else {
                    const int b = m / Tc, t = m % Tc;
                    const int which = n / HIDc;
                    const int h = (n % HIDc) / HDc;
                    const int d = n % HDc;
                    if (which == 0) {
                        qb[(((size_t)b * NHc + h) * Tc + t) * HDc + d] = (__bf16)val;
                    } else if (which == 1) {
                        kb[(((size_t)b * NHc + h) * Tc + t) * HDc + d] = (__bf16)val;
                    } else {
                        vt[(((size_t)b * NHc + h) * HDc + d) * Tc + t] = (__bf16)val;
                    }
                }
            }
        }
    }
}

// ---------------------------------------------------------------------------
// RoPE in-place on qb/kb (bf16). q additionally pre-scaled by 1/sqrt(HD).
// ---------------------------------------------------------------------------
__global__ __launch_bounds__(256)
void rope_kernel(__bf16* __restrict__ qb, __bf16* __restrict__ kb,
                 const float* __restrict__ cosT, const float* __restrict__ sinT) {
    const int i = blockIdx.x * blockDim.x + threadIdx.x;
    const int j = i & 63;            // pair index within head dim
    const int t = (i >> 6) & (Tc - 1);
    const size_t e0 = (size_t)i * 2; // even element, odd = e0+1
    const float c = cosT[t * 64 + j];
    const float s = sinT[t * 64 + j];

    float qe = (float)qb[e0], qo = (float)qb[e0 + 1];
    qb[e0]     = (__bf16)((qe * c - qo * s) * QSCALE);
    qb[e0 + 1] = (__bf16)((qe * s + qo * c) * QSCALE);

    float ke = (float)kb[e0], ko = (float)kb[e0 + 1];
    kb[e0]     = (__bf16)(ke * c - ko * s);
    kb[e0 + 1] = (__bf16)(ke * s + ko * c);
}

// ---------------------------------------------------------------------------
// Flash attention: block = 8 waves, all on the SAME (b,h); each wave owns a
// 16-row Q tile. K/V 32-key tiles staged once per block into LDS via
// global_load_async_to_lds_b128 (ASYNCcnt), then consumed by all waves.
//
// LDS carve (single segment, byte offsets used by the async asm):
//   [0      .. 8191 ]  Kst[32 keys][128 d]   bf16
//   [8192   .. 16383]  Vts[128 d ][32 keys]  bf16
//   [16384  .. 24575]  Pst[8 waves][16][32]  bf16
// ---------------------------------------------------------------------------
static constexpr unsigned KST_OFF = 0;        // bf16 element offsets
static constexpr unsigned VTS_OFF = 4096;
static constexpr unsigned PST_OFF = 8192;

__global__ __launch_bounds__(256)
void flash_attn_wmma(const __bf16* __restrict__ qb, const __bf16* __restrict__ kb,
                     const __bf16* __restrict__ vt, float* __restrict__ attn) {
    __shared__ alignas(16) __bf16 smem[12288];   // 24 KB

    const int tid  = threadIdx.x;
    const int lane = tid & 31;
    const int wid  = tid >> 5;
    const int half = lane >> 4;
    const int lm   = lane & 15;

    // 16 blocks per (b,h); 8 q-tiles (128 q rows) per block.
    const int bh   = blockIdx.x >> 4;
    const int qblk = blockIdx.x & 15;
    const int b    = bh / NHc, h = bh % NHc;
    const int q0   = (qblk * 8 + wid) * 16;

    const __bf16* Q  = qb + (size_t)bh * Tc * HDc;
    const __bf16* Kp = kb + (size_t)bh * Tc * HDc;
    const __bf16* Vp = vt + (size_t)bh * HDc * Tc;

    // ---- Q fragments: 4 x (16x32 bf16 A-matrix), K-dim = head dim ----
    v16bf qf[4];
#pragma unroll
    for (int kc = 0; kc < 4; ++kc) {
        const __bf16* src = Q + (size_t)(q0 + lm) * HDc + kc * 32;
        bf16x8 lo = *(const bf16x8*)(src + half * 8);
        bf16x8 hi = *(const bf16x8*)(src + 16 + half * 8);
        qf[kc] = pack16(lo, hi);
    }

    // Per-thread staging coordinates (16 bf16 = 32 B each, i.e. 2 async b128):
    //   K tile: 32 rows(keys) x 128 cols(d):  row = tid/8,  col = (tid%8)*16
    //   V tile: 128 rows(d)  x 32 cols(keys): row = tid/2,  col = (tid%2)*16
    const int krow = tid >> 3, kcol = (tid & 7) * 16;
    const int vrow = tid >> 1, vcol = (tid & 1) * 16;
    const unsigned klds = (KST_OFF + (unsigned)(krow * 128 + kcol)) * 2u;
    const unsigned vlds = (VTS_OFF + (unsigned)(vrow * 32 + vcol)) * 2u;

    v8f O[8] = {};
    float mrow[8], lrow[8];
#pragma unroll
    for (int r = 0; r < 8; ++r) { mrow[r] = -1e30f; lrow[r] = 0.0f; }

    for (int j = 0; j < Tc; j += 32) {
        // ---- async stage K and V 32-key tiles into LDS (block-wide) ----
        const __bf16* kg = Kp + (size_t)(j + krow) * HDc + kcol;
        async_ld16(klds,      kg);
        async_ld16(klds + 16, kg + 8);
        const __bf16* vg = Vp + (size_t)vrow * Tc + j + vcol;
        async_ld16(vlds,      vg);
        async_ld16(vlds + 16, vg + 8);
        async_wait0();
        __syncthreads();

        // ---- scores S = Q * K^T for 32 keys (two 16x16 accum tiles) ----
        v8f S0 = {}, S1 = {};
#pragma unroll
        for (int kc = 0; kc < 4; ++kc) {
            const __bf16* k0 = smem + KST_OFF + (unsigned)lm * 128 + kc * 32 + half * 8;
            v16bf kf0 = pack16(*(const bf16x8*)k0, *(const bf16x8*)(k0 + 16));
            S0 = __builtin_amdgcn_wmma_f32_16x16x32_bf16(false, qf[kc], false, kf0,
                                                         (short)0, S0, false, false);
            const __bf16* k1 = k0 + 16 * 128;
            v16bf kf1 = pack16(*(const bf16x8*)k1, *(const bf16x8*)(k1 + 16));
            S1 = __builtin_amdgcn_wmma_f32_16x16x32_bf16(false, qf[kc], false, kf1,
                                                         (short)0, S1, false, false);
        }

        // ---- online softmax update; stage P into wave-private LDS region ----
        float corr[8];
#pragma unroll
        for (int r = 0; r < 8; ++r) {
            float mx = lane16_max(fmaxf(S0[r], S1[r]));
            float mnew = fmaxf(mrow[r], mx);
            corr[r] = __expf(mrow[r] - mnew);
            mrow[r] = mnew;
            float p0 = __expf(S0[r] - mnew);
            float p1 = __expf(S1[r] - mnew);
            __bf16* prow = smem + PST_OFF + wid * 512 + (8 * half + r) * 32;
            prow[lm]      = (__bf16)p0;
            prow[16 + lm] = (__bf16)p1;
            float rs = lane16_sum(p0 + p1);
            lrow[r] = lrow[r] * corr[r] + rs;
        }
#pragma unroll
        for (int c = 0; c < 8; ++c)
#pragma unroll
            for (int r = 0; r < 8; ++r) O[c][r] *= corr[r];

        // ---- P as A-fragment (same-wave LDS ops are in order) ----
        const __bf16* pp = smem + PST_OFF + wid * 512 + lm * 32;
        v16bf pf = pack16(*(const bf16x8*)(pp + half * 8),
                          *(const bf16x8*)(pp + 16 + half * 8));

        // ---- O += P * V : 8 d-chunks of 16 cols, V from LDS ----
#pragma unroll
        for (int c = 0; c < 8; ++c) {
            const __bf16* vb = smem + VTS_OFF + (unsigned)(c * 16 + lm) * 32 + half * 8;
            v16bf vf = pack16(*(const bf16x8*)vb, *(const bf16x8*)(vb + 16));
            O[c] = __builtin_amdgcn_wmma_f32_16x16x32_bf16(false, pf, false, vf,
                                                           (short)0, O[c], false, false);
        }
        __syncthreads();   // protect K/V LDS before next tile's staging
    }

    // ---- normalize and store: attn[b, q, h*HD + d] ----
    float inv[8];
#pragma unroll
    for (int r = 0; r < 8; ++r) inv[r] = 1.0f / lrow[r];
#pragma unroll
    for (int c = 0; c < 8; ++c) {
#pragma unroll
        for (int r = 0; r < 8; ++r) {
            const int qrow = q0 + 8 * half + r;
            const int d = c * 16 + lm;
            attn[((size_t)b * Tc + qrow) * HIDc + h * HDc + d] = O[c][r] * inv[r];
        }
    }
}

// ---------------------------------------------------------------------------
extern "C" void kernel_launch(void* const* d_in, const int* in_sizes, int n_in,
                              void* d_out, int out_size, void* d_ws, size_t ws_size,
                              hipStream_t stream) {
    const float* x      = (const float*)d_in[0];
    const float* w_qkv  = (const float*)d_in[1];
    const float* w_proj = (const float*)d_in[2];
    const float* b_proj = (const float*)d_in[3];
    const float* cosT   = (const float*)d_in[4];
    const float* sinT   = (const float*)d_in[5];
    float* out = (float*)d_out;

    const size_t qkN = (size_t)Bc * NHc * Tc * HDc; // 16,777,216 elements
    char* p = (char*)d_ws;
    __bf16* qb = (__bf16*)p; p += qkN * sizeof(__bf16);
    __bf16* kb = (__bf16*)p; p += qkN * sizeof(__bf16);
    __bf16* vt = (__bf16*)p; p += qkN * sizeof(__bf16);
    float* attnb = (float*)p;                        // [B, T, HID] f32

    dim3 blk(256);

    // 1) QKV projection -> bf16 qb/kb + transposed vt
    dim3 g1((3 * HIDc) / 128, Mrows / 64);
    gemm_bf16_wmma<1><<<g1, blk, 0, stream>>>(x, w_qkv, nullptr, nullptr,
                                              qb, kb, vt, Mrows, 3 * HIDc, HIDc);

    // 2) RoPE on q and k (q pre-scaled by 1/sqrt(HD))
    const int rope_threads = Bc * NHc * Tc * (HDc / 2); // 8,388,608
    rope_kernel<<<rope_threads / 256, blk, 0, stream>>>(qb, kb, cosT, sinT);

    // 3) Flash attention (8 waves/block sharing one (b,h); async K/V staging)
    const int qtiles = Bc * NHc * (Tc / 16);            // 8192
    flash_attn_wmma<<<qtiles / 8, blk, 0, stream>>>(qb, kb, vt, attnb);

    // 4) Output projection + bias
    dim3 g4(HIDc / 128, Mrows / 64);
    gemm_bf16_wmma<0><<<g4, blk, 0, stream>>>(attnb, w_proj, b_proj, out,
                                              nullptr, nullptr, nullptr,
                                              Mrows, HIDc, HIDc);
}
